// GraphSAGE_90512140795978
// MI455X (gfx1250) — matrix operand order repaired and checked by reading, not compiled, not run on hardware
//
#include <hip/hip_runtime.h>

typedef __attribute__((ext_vector_type(16))) _Float16 v16h;
typedef __attribute__((ext_vector_type(8)))  float    v8f;

#define N_NODES 100000
#define N_EDGES 1600000
#define M_TILES 6250   // 100000 / 16 exactly
#define MPB     5      // M-tiles per block; 6250 = 5 * 1250

// ---------------- utility kernels ----------------

__global__ void k_fill_zero(float* __restrict__ p, int n) {
  int i = blockIdx.x * blockDim.x + threadIdx.x;
  if (i < n) p[i] = 0.f;
}

// Convert f32 weight [K][N] into pre-swizzled f16 B-fragments matching the
// wave32 WMMA B operand in the contiguous k order the A side uses: frag index = ((ntile*KT + kt)*32 + lane)*16 + j
// with k = kt*32 + (lane>>4)*16 + j, n = ntile*16 + (lane&15). Zero-pad OOB.
__global__ void k_cvt_swz(const float* __restrict__ src, _Float16* __restrict__ dst,
                          int K, int N, int KT, int NTt) {
  int i = blockIdx.x * blockDim.x + threadIdx.x;
  int total = NTt * KT * 32 * 16;
  if (i >= total) return;
  int j     = i & 15;
  int lane  = (i >> 4) & 31;
  int ktn   = i >> 9;            // ntile*KT + kt
  int kt    = ktn % KT;
  int ntile = ktn / KT;
  int k  = kt * 32 + (lane >> 4) * 16 + j;
  int nn = ntile * 16 + (lane & 15);
  dst[i] = (k < K && nn < N) ? (_Float16)src[k * N + nn] : (_Float16)0.f;
}

__global__ void k_degree(const int* __restrict__ dst, float* __restrict__ deg) {
  int e = blockIdx.x * blockDim.x + threadIdx.x;
  if (e < N_EDGES) unsafeAtomicAdd(&deg[dst[e]], 1.0f);
}

// agg[dst] += w * h[src], 4 features per thread (float4 gather, 4 fp32 atomics)
template <int C>
__global__ void k_scatter(const int* __restrict__ src, const int* __restrict__ dst,
                          const float* __restrict__ ew, const float* __restrict__ h,
                          float* __restrict__ agg, int useW) {
  constexpr int NCHUNK = C >> 2;
  int tid = blockIdx.x * blockDim.x + threadIdx.x;
  if (tid >= N_EDGES * NCHUNK) return;
  int e = tid / NCHUNK;
  int c = (tid - e * NCHUNK) << 2;
  int s = src[e], d = dst[e];
  float w = useW ? ew[e] : 1.0f;
  const float4 v = *(const float4*)(h + (size_t)s * C + c);
  float* a = agg + (size_t)d * C + c;
  unsafeAtomicAdd(a + 0, v.x * w);
  unsafeAtomicAdd(a + 1, v.y * w);
  unsafeAtomicAdd(a + 2, v.z * w);
  unsafeAtomicAdd(a + 3, v.w * w);
}

// ---------------- WMMA GEMM kernels ----------------
// Fragment layouts (wave32, ISA 7.12.2):
//  A f16 16x32: lane l -> row m=l&15, half=l>>4; elem j -> k = half*16 + j (contiguous order chosen here, not the ISA table's)
//  B f16 32x16: pre-swizzled in memory -> one 32B contiguous chunk per lane
//  C/D f32 16x16: lane l -> col n=l&15; vgpr i -> row m = (l>>4)*8 + i

// Fused SAGE layer GEMM + epilogue:
//   t = (agg/max(deg,1)) @ Wl + h @ Wr + b ;  out = (relu?) t / max(||t||2, 1e-12)
template <int KP, int NT, int C, int NOUT, int OSTRIDE, int RELU>
__global__ __launch_bounds__(NT * 32)
void k_sage_gemm(const float* __restrict__ agg, const float* __restrict__ deg,
                 const float* __restrict__ h,
                 const _Float16* __restrict__ wlf, const _Float16* __restrict__ wrf,
                 const float* __restrict__ bias, float* __restrict__ out) {
  constexpr int KT  = KP / 32;
  constexpr int NP  = NT * 16;
  constexpr int NTH = NT * 32;
  constexpr int KG  = KP / 4;          // float4 groups per row
  __shared__ _Float16 lA[16][KP];      // (agg/deg) tile, f16
  __shared__ _Float16 lH[16][KP];      // h tile, f16
  __shared__ float    lO[16][NP + 1];  // output staging (+1: conflict-free row sums)
  __shared__ float    lS[16];          // per-row 1/norm

  const int tid   = threadIdx.x;
  const int lane  = tid & 31;
  const int ntile = tid >> 5;
  const int mrow  = lane & 15;
  const int half  = lane >> 4;
  const int n     = ntile * 16 + mrow;

  // B fragments: one 32-byte vector load per (matrix, ktile); live in VGPRs
  v16h bl[KT], br[KT];
  #pragma unroll
  for (int kt = 0; kt < KT; ++kt) {
    bl[kt] = *(const v16h*)(wlf + (((ntile * KT) + kt) * 32 + lane) * 16);
    br[kt] = *(const v16h*)(wrf + (((ntile * KT) + kt) * 32 + lane) * 16);
  }
  const float bn = (n < NOUT) ? bias[n] : 0.f;

  for (int mt = 0; mt < MPB; ++mt) {
    const int mtile = blockIdx.x * MPB + mt;
    // stage A tiles: f32 global (float4) -> f16 LDS (merged ds_store_b64)
    for (int g = tid; g < 16 * KG; g += NTH) {
      int m  = g / KG;
      int k4 = (g - m * KG) * 4;
      int gm = mtile * 16 + m;
      float4 va = {0.f, 0.f, 0.f, 0.f}, vh = va;
      if (k4 < C) {  // C % 4 == 0, so whole group in/out of range together
        float invd = 1.0f / fmaxf(deg[gm], 1.0f);
        va = *(const float4*)(agg + (size_t)gm * C + k4);
        vh = *(const float4*)(h + (size_t)gm * C + k4);
        va.x *= invd; va.y *= invd; va.z *= invd; va.w *= invd;
      }
      lA[m][k4 + 0] = (_Float16)va.x; lA[m][k4 + 1] = (_Float16)va.y;
      lA[m][k4 + 2] = (_Float16)va.z; lA[m][k4 + 3] = (_Float16)va.w;
      lH[m][k4 + 0] = (_Float16)vh.x; lH[m][k4 + 1] = (_Float16)vh.y;
      lH[m][k4 + 2] = (_Float16)vh.z; lH[m][k4 + 3] = (_Float16)vh.w;
    }
    __syncthreads();

    // preload all A fragments, then run the WMMA chain
    v16h a1[KT], a2[KT];
    #pragma unroll
    for (int kt = 0; kt < KT; ++kt) {
      #pragma unroll
      for (int j = 0; j < 16; ++j) {
        int ka = kt * 32 + half * 16 + j;
        a1[kt][j] = lA[mrow][ka];
        a2[kt][j] = lH[mrow][ka];
      }
    }
    v8f acc = {};
    #pragma unroll
    for (int kt = 0; kt < KT; ++kt) {
      acc = __builtin_amdgcn_wmma_f32_16x16x32_f16(false, a1[kt], false, bl[kt], (short)0, acc, false, false);
      acc = __builtin_amdgcn_wmma_f32_16x16x32_f16(false, a2[kt], false, br[kt], (short)0, acc, false, false);
    }

    // epilogue: bias -> LDS, per-row L2 norm, scale (+relu), store
    #pragma unroll
    for (int i = 0; i < 8; ++i)
      lO[half * 8 + i][n] = acc[i] + bn;
    __syncthreads();

    if (tid < 16) {
      float ss = 0.f;
      for (int c = 0; c < NP; ++c) { float v = lO[tid][c]; ss += v * v; }
      lS[tid] = 1.0f / fmaxf(sqrtf(ss), 1e-12f);
    }
    __syncthreads();

    float* op = out + (size_t)(mtile * 16) * OSTRIDE;
    for (int i = tid; i < 16 * NOUT; i += NTH) {
      int m = i / NOUT, c = i - m * NOUT;
      float v = lO[m][c] * lS[m];
      if (RELU) v = fmaxf(v, 0.f);
      op[(size_t)m * OSTRIDE + c] = v;
    }
    __syncthreads();
  }
}

// h0 = relu(x @ Wemb + b); N == NT*16 exactly (80)
template <int KP, int NT, int K, int N>
__global__ __launch_bounds__(NT * 32)
void k_embed_gemm(const float* __restrict__ x,
                  const _Float16* __restrict__ wf, const float* __restrict__ bias,
                  float* __restrict__ hout) {
  constexpr int KT  = KP / 32;
  constexpr int NTH = NT * 32;
  constexpr int KG  = KP / 4;
  __shared__ _Float16 lA[16][KP];
  const int tid   = threadIdx.x;
  const int lane  = tid & 31;
  const int ntile = tid >> 5;
  const int mrow  = lane & 15;
  const int half  = lane >> 4;
  const int n     = ntile * 16 + mrow;

  v16h bw[KT];
  #pragma unroll
  for (int kt = 0; kt < KT; ++kt)
    bw[kt] = *(const v16h*)(wf + (((ntile * KT) + kt) * 32 + lane) * 16);
  const float bn = bias[n];

  for (int mt = 0; mt < MPB; ++mt) {
    const int mtile = blockIdx.x * MPB + mt;
    for (int g = tid; g < 16 * KG; g += NTH) {
      int m  = g / KG;
      int k4 = (g - m * KG) * 4;
      int gm = mtile * 16 + m;
      float4 v = {0.f, 0.f, 0.f, 0.f};
      if (k4 < K)  // K % 4 == 0
        v = *(const float4*)(x + (size_t)gm * K + k4);
      lA[m][k4 + 0] = (_Float16)v.x; lA[m][k4 + 1] = (_Float16)v.y;
      lA[m][k4 + 2] = (_Float16)v.z; lA[m][k4 + 3] = (_Float16)v.w;
    }
    __syncthreads();

    v16h a[KT];
    #pragma unroll
    for (int kt = 0; kt < KT; ++kt) {
      #pragma unroll
      for (int j = 0; j < 16; ++j)
        a[kt][j] = lA[mrow][kt * 32 + half * 16 + j];
    }
    v8f acc = {};
    #pragma unroll
    for (int kt = 0; kt < KT; ++kt)
      acc = __builtin_amdgcn_wmma_f32_16x16x32_f16(false, a[kt], false, bw[kt], (short)0, acc, false, false);

    // 8 stores off one base address with constant i*N offsets
    float* p = hout + (size_t)(mtile * 16 + half * 8) * N + n;
    #pragma unroll
    for (int i = 0; i < 8; ++i)
      p[i * N] = fmaxf(acc[i] + bn, 0.f);
    __syncthreads();
  }
}

// ---------------- launcher ----------------

extern "C" void kernel_launch(void* const* d_in, const int* in_sizes, int n_in,
                              void* d_out, int out_size, void* d_ws, size_t ws_size,
                              hipStream_t stream) {
  // inputs per setup_inputs() order
  const float* x      = (const float*)d_in[0];
  const int*   ei     = (const int*)d_in[1];          // [2, E]
  const float* ew     = (const float*)d_in[2];
  const float* emb_W  = (const float*)d_in[3];
  const float* emb_b  = (const float*)d_in[4];
  const float* Wl[4]  = {(const float*)d_in[5],  (const float*)d_in[8],
                         (const float*)d_in[11], (const float*)d_in[14]};
  const float* Wr[4]  = {(const float*)d_in[6],  (const float*)d_in[9],
                         (const float*)d_in[12], (const float*)d_in[15]};
  const float* bb[4]  = {(const float*)d_in[7],  (const float*)d_in[10],
                         (const float*)d_in[13], (const float*)d_in[16]};
  const float* Wlo    = (const float*)d_in[17];
  const float* Wro    = (const float*)d_in[18];
  const float* bo     = (const float*)d_in[19];
  const int* srcI = ei;
  const int* dstI = ei + N_EDGES;
  float* outp = (float*)d_out;

  // workspace carve-out
  char* ws = (char*)d_ws;
  size_t off = 0;
  auto alloc = [&](size_t bytes) -> void* {
    void* p = ws + off;
    off += (bytes + 255) & ~(size_t)255;
    return p;
  };
  float* h0  = (float*)alloc((size_t)N_NODES * 80 * 4);
  float* hb  = (float*)alloc((size_t)N_NODES * 64 * 4);
  float* hc  = (float*)alloc((size_t)N_NODES * 64 * 4);
  float* agg = (float*)alloc((size_t)N_NODES * 80 * 4);
  float* deg = (float*)alloc((size_t)N_NODES * 4);
  // pre-swizzled f16 weight fragments: NT*KT*32*16 halfs each
  _Float16* wemb16 = (_Float16*)alloc((size_t)5 * 4 * 512 * 2);
  _Float16* wl16[4], *wr16[4];
  wl16[0] = (_Float16*)alloc((size_t)4 * 3 * 512 * 2);
  wr16[0] = (_Float16*)alloc((size_t)4 * 3 * 512 * 2);
  for (int i = 1; i < 4; ++i) {
    wl16[i] = (_Float16*)alloc((size_t)4 * 2 * 512 * 2);
    wr16[i] = (_Float16*)alloc((size_t)4 * 2 * 512 * 2);
  }
  _Float16* wlo16 = (_Float16*)alloc((size_t)2 * 2 * 512 * 2);
  _Float16* wro16 = (_Float16*)alloc((size_t)2 * 2 * 512 * 2);

  auto cvt = [&](const float* s, _Float16* d, int K, int N, int KT, int NTt) {
    int n = NTt * KT * 512;
    k_cvt_swz<<<(n + 255) / 256, 256, 0, stream>>>(s, d, K, N, KT, NTt);
  };
  auto zero = [&](float* p, int n) {
    k_fill_zero<<<(n + 255) / 256, 256, 0, stream>>>(p, n);
  };

  // weight conversion + swizzle (f32 [K][N] -> f16 B-fragments)
  cvt(emb_W, wemb16, 100, 80, 4, 5);
  cvt(Wl[0], wl16[0], 80, 64, 3, 4);
  cvt(Wr[0], wr16[0], 80, 64, 3, 4);
  for (int i = 1; i < 4; ++i) {
    cvt(Wl[i], wl16[i], 64, 64, 2, 4);
    cvt(Wr[i], wr16[i], 64, 64, 2, 4);
  }
  cvt(Wlo, wlo16, 64, 18, 2, 2);
  cvt(Wro, wro16, 64, 18, 2, 2);

  // degrees (shared by all 5 conv layers: same dst index set)
  zero(deg, N_NODES);
  k_degree<<<(N_EDGES + 255) / 256, 256, 0, stream>>>(dstI, deg);

  const int GB = M_TILES / MPB;  // 1250 blocks

  // embedding: h0 = relu(x @ Wemb + b)
  k_embed_gemm<128, 5, 100, 80><<<GB, 5 * 32, 0, stream>>>(x, wemb16, emb_b, h0);

  // layer 0: 80 -> 64, h0 -> hb
  {
    zero(agg, N_NODES * 80);
    int nthr = N_EDGES * (80 >> 2);
    k_scatter<80><<<(nthr + 255) / 256, 256, 0, stream>>>(srcI, dstI, ew, h0, agg, 1);
    k_sage_gemm<96, 4, 80, 64, 64, 1><<<GB, 4 * 32, 0, stream>>>(agg, deg, h0,
                                                                 wl16[0], wr16[0], bb[0], hb);
  }

  // layers 1..3: 64 -> 64, ping-pong hb/hc
  float* cur = hb;
  float* nxt = hc;
  for (int L = 1; L < 4; ++L) {
    zero(agg, N_NODES * 64);
    int nthr = N_EDGES * (64 >> 2);
    k_scatter<64><<<(nthr + 255) / 256, 256, 0, stream>>>(srcI, dstI, ew, cur, agg, 1);
    k_sage_gemm<64, 4, 64, 64, 64, 1><<<GB, 4 * 32, 0, stream>>>(agg, deg, cur,
                                                                 wl16[L], wr16[L], bb[L], nxt);
    float* t = cur; cur = nxt; nxt = t;
  }

  // output layer: 64 -> 18 (N padded to 32), ones edge weights, no ReLU
  {
    zero(agg, N_NODES * 64);
    int nthr = N_EDGES * (64 >> 2);
    k_scatter<64><<<(nthr + 255) / 256, 256, 0, stream>>>(srcI, dstI, ew, cur, agg, 0);
    k_sage_gemm<64, 2, 64, 18, 18, 0><<<GB, 2 * 32, 0, stream>>>(agg, deg, cur,
                                                                 wlo16, wro16, bo, outp);
  }
}
